// _VectorQuantizer_22514218565705
// MI455X (gfx1250) — matrix operand order repaired and checked by reading, not compiled, not run on hardware
//
#include <hip/hip_runtime.h>
#include <hip/hip_bf16.h>

// ---- problem constants (from reference) ----
#define BATCH   16
#define DIM     256
#define FRAMES  2048
#define KCB     1024
#define NVEC    (BATCH * FRAMES)          // 32768 query vectors
#define LDS_ROW 264                       // 256 + 8 halves pad: 132 dwords == 4 mod 64 -> conflict-light b128 reads

typedef _Float16 h16;
typedef __attribute__((ext_vector_type(16))) _Float16 v16h;
typedef __attribute__((ext_vector_type(8)))  _Float16 v8h;
typedef __attribute__((ext_vector_type(8)))  float    v8f;
typedef __attribute__((__vector_size__(16))) int      vi4;   // matches builtin param pointee

#define AS1 __attribute__((address_space(1)))
#define AS3 __attribute__((address_space(3)))

#if __has_builtin(__builtin_amdgcn_global_load_async_to_lds_b128)
#define VQ_ASYNC_COPY 1
#endif

template <int N> __device__ __forceinline__ void wait_asynccnt() {
#if __has_builtin(__builtin_amdgcn_s_wait_asynccnt)
  __builtin_amdgcn_s_wait_asynccnt(N);
#else
  asm volatile("s_wait_asynccnt %0" ::"n"(N) : "memory");
#endif
}

// ---------------------------------------------------------------------------
// Prep: codebook fp32 -> f16 (WMMA B operand), c_sq[k] = ||c_k||^2, zero loss.
// grid = 1024 (one block per codebook row), block = 256 (one thread per dim)
// ---------------------------------------------------------------------------
__global__ __launch_bounds__(256) void vq_prep(const float* __restrict__ cb,
                                               h16* __restrict__ cb16,
                                               float* __restrict__ csq,
                                               float* __restrict__ loss_slot) {
  const int row = blockIdx.x;
  const int tid = threadIdx.x;
  float v = cb[(size_t)row * DIM + tid];
  cb16[(size_t)row * DIM + tid] = (h16)v;
  float s = v * v;
#pragma unroll
  for (int mask = 16; mask >= 1; mask >>= 1) s += __shfl_xor(s, mask, 32);
  __shared__ float red[8];
  if ((tid & 31) == 0) red[tid >> 5] = s;
  __syncthreads();
  if (tid == 0) {
    float tot = 0.f;
#pragma unroll
    for (int i = 0; i < 8; ++i) tot += red[i];
    csq[row] = tot;
    if (row == 0) *loss_slot = 0.f;   // init atomic accumulator
  }
}

// ---------------------------------------------------------------------------
// Main: per wave 16 frames; loop 64 codebook tiles of 16 entries.
// scores via v_wmma_f32_16x16x32_f16; argmin tracked in fp32.
// Codebook tiles staged into LDS via GLOBAL_LOAD_ASYNC_TO_LDS (ASYNCcnt),
// double-buffered, with issue-ahead of one tile.
// grid = 256 blocks x 256 threads (8 waves, 128 frames / block)
// ---------------------------------------------------------------------------
__global__ __launch_bounds__(256) void vq_main(const float* __restrict__ x,
                                               const float* __restrict__ cb,
                                               const h16*   __restrict__ cb16,
                                               const float* __restrict__ csq,
                                               float* __restrict__ outq,
                                               float* __restrict__ codes_out,
                                               float* __restrict__ loss_out) {
#if VQ_ASYNC_COPY
  __shared__ __align__(16) h16 tile[2][16 * LDS_ROW];  // ping-pong codebook tiles (f16)
#else
  __shared__ __align__(16) h16 tile[1][16 * LDS_ROW];
#endif
  __shared__ int codesLDS[8][16];

  const int tid  = threadIdx.x;
  const int wave = tid >> 5;
  const int lane = tid & 31;
  const int m    = lane & 15;        // row (frame) / column (cb entry) within 16-tile
  const int hi   = lane >> 4;        // half-wave selector per WMMA layout

  const int gf0 = blockIdx.x * 128 + wave * 16;   // global flat frame base for this wave
  const int b   = gf0 >> 11;                      // / FRAMES
  const int f0  = gf0 & (FRAMES - 1);

  // ---- build A fragments (16 frames x 256 dims, f16) in registers; x_sq on the side ----
  // A 16x32 f16 layout: lane(m,hi): v16h elems 0..7 = K 8hi+0..7, elems 8..15 = K 16+8hi+0..7
  const float* xb = x + (size_t)b * DIM * FRAMES + f0 + m;
  v16h a[8];
  float xsq = 0.f;
#pragma unroll
  for (int kk = 0; kk < 8; ++kk) {
    const int d0 = kk * 32 + hi * 8;
#pragma unroll
    for (int j = 0; j < 8; ++j) {
      float lo = xb[(size_t)(d0 + j) * FRAMES];
      float hh = xb[(size_t)(d0 + 16 + j) * FRAMES];
      a[kk][j]     = (h16)lo;
      a[kk][8 + j] = (h16)hh;
      xsq += lo * lo + hh * hh;
    }
  }
  xsq += __shfl_xor(xsq, 16, 32);          // combine the two half-lane dim partitions
  float xsqr[8];                           // xsqr[r] = ||x_{8hi+r}||^2
#pragma unroll
  for (int r = 0; r < 8; ++r) xsqr[r] = __shfl(xsq, hi * 8 + r, 32);

  float best[8];
  int   bidx[8];
#pragma unroll
  for (int r = 0; r < 8; ++r) { best[r] = 3.4e38f; bidx[r] = 0; }

  // thread -> two 16B chunks of the 8KB tile
  const int q0 = tid, q1 = tid + 256;
  const int r0 = q0 >> 5, o0 = q0 & 31;    // row 0..15, 16B chunk 0..31
  const int r1 = q1 >> 5, o1 = q1 & 31;

#if VQ_ASYNC_COPY
  // ---- async staging: global -> LDS without VGPR round trip (ASYNCcnt) ----
  {
    __builtin_amdgcn_global_load_async_to_lds_b128(
        (AS1 vi4*)(cb16 + ((size_t)r0 << 8) + o0 * 8),
        (AS3 vi4*)&tile[0][r0 * LDS_ROW + o0 * 8], 0, 0);
    __builtin_amdgcn_global_load_async_to_lds_b128(
        (AS1 vi4*)(cb16 + ((size_t)r1 << 8) + o1 * 8),
        (AS3 vi4*)&tile[0][r1 * LDS_ROW + o1 * 8], 0, 0);
  }
  for (int t = 0; t < 64; ++t) {
    const h16* lt = tile[t & 1];
    if (t < 63) {                                      // issue tile t+1 into other buffer
      const int nb = (t + 1) & 1;
      __builtin_amdgcn_global_load_async_to_lds_b128(
          (AS1 vi4*)(cb16 + (((size_t)(t + 1) * 16 + r0) << 8) + o0 * 8),
          (AS3 vi4*)&tile[nb][r0 * LDS_ROW + o0 * 8], 0, 0);
      __builtin_amdgcn_global_load_async_to_lds_b128(
          (AS1 vi4*)(cb16 + (((size_t)(t + 1) * 16 + r1) << 8) + o1 * 8),
          (AS3 vi4*)&tile[nb][r1 * LDS_ROW + o1 * 8], 0, 0);
      wait_asynccnt<2>();                              // tile t's 2 ops retired (in-order)
    } else {
      wait_asynccnt<0>();
    }
    __syncthreads();                                   // tile t visible to all waves
#else
  // ---- fallback: register-staged tiles with issue-ahead prefetch ----
  uint4 s0 = *(const uint4*)(cb16 + ((size_t)r0 << 8) + o0 * 8);
  uint4 s1 = *(const uint4*)(cb16 + ((size_t)r1 << 8) + o1 * 8);
  for (int t = 0; t < 64; ++t) {
    const h16* lt = tile[0];
    __syncthreads();                                   // previous tile fully consumed
    *(uint4*)&tile[0][r0 * LDS_ROW + o0 * 8] = s0;
    *(uint4*)&tile[0][r1 * LDS_ROW + o1 * 8] = s1;
    if (t < 63) {                                      // prefetch next tile (overlaps WMMA)
      s0 = *(const uint4*)(cb16 + (((size_t)(t + 1) * 16 + r0) << 8) + o0 * 8);
      s1 = *(const uint4*)(cb16 + (((size_t)(t + 1) * 16 + r1) << 8) + o1 * 8);
    }
    __syncthreads();
#endif

    const int n = t * 16 + m;                          // this lane's codebook column
    float cs = csq[n];
    v8f c = {};
    // B 32x16 f16 layout: lane(n=m,hi): v16h elems 0..7 = K 8hi+0..7, elems 8..15 = K 16+8hi+0..7
    // of column n (row n of the row-major codebook).
#pragma unroll
    for (int kk = 0; kk < 8; ++kk) {
      const v8h* lp = (const v8h*)&lt[m * LDS_ROW + kk * 32 + hi * 8];
      v8h b0 = lp[0], b1 = lp[2];
      v16h bf = __builtin_shufflevector(b0, b1, 0, 1, 2, 3, 4, 5, 6, 7,
                                        8, 9, 10, 11, 12, 13, 14, 15);
      c = __builtin_amdgcn_wmma_f32_16x16x32_f16(false, a[kk], false, bf,
                                                 (short)0, c, false, false);
    }
    // C layout: lane(n=m,hi) holds rows M = 8hi + r in c[r]
#pragma unroll
    for (int r = 0; r < 8; ++r) {
      float score = cs - 2.0f * c[r];                  // x_sq omitted: constant per row
      if (score < best[r]) { best[r] = score; bidx[r] = n; }
    }
    __syncthreads();                                   // all waves done with tile t
  }

  // ---- argmin across the 16 columns held by the 16 lanes of each half ----
#pragma unroll
  for (int r = 0; r < 8; ++r) {
#pragma unroll
    for (int mask = 1; mask <= 8; mask <<= 1) {
      float od = __shfl_xor(best[r], mask, 32);
      int   oi = __shfl_xor(bidx[r], mask, 32);
      if (od < best[r] || (od == best[r] && oi < bidx[r])) { best[r] = od; bidx[r] = oi; }
    }
  }

  // ---- writers: codes + loss partial (rows m = 8hi + r) ----
  if (m == 0) {
    float lp = 0.f;
#pragma unroll
    for (int r = 0; r < 8; ++r) {
      const int gm = hi * 8 + r;
      codesLDS[wave][gm] = bidx[r];
      codes_out[gf0 + gm] = (float)bidx[r];
      lp += xsqr[r] + best[r];                          // = min squared distance
    }
    atomicAdd(loss_out, lp * (1.25f / ((float)NVEC * (float)DIM)));
  }
  __syncthreads();

  // ---- gather quantized output (fp32 codebook, frame-contiguous 64B stores) ----
  const int code_m = codesLDS[wave][m];
  const float* crow = cb + (size_t)code_m * DIM;
  float* ob = outq + (size_t)b * DIM * FRAMES + f0 + m;
#pragma unroll 4
  for (int d0 = 0; d0 < DIM; d0 += 2) {
    const int d = d0 + hi;
    ob[(size_t)d * FRAMES] = crow[d];
  }
}

// ---------------------------------------------------------------------------
extern "C" void kernel_launch(void* const* d_in, const int* in_sizes, int n_in,
                              void* d_out, int out_size, void* d_ws, size_t ws_size,
                              hipStream_t stream) {
  const float* x  = (const float*)d_in[0];   // [16, 256, 2048] fp32
  const float* cb = (const float*)d_in[1];   // [1024, 256] fp32

  h16*   cb16 = (h16*)d_ws;                                    // 512 KB f16 codebook
  float* csq  = (float*)((char*)d_ws + (size_t)KCB * DIM * 2); // 4 KB ||c||^2

  float* outq  = (float*)d_out;                              // [16,256,2048]
  float* codes = outq + (size_t)BATCH * DIM * FRAMES;        // [16,2048]
  float* loss  = codes + (size_t)BATCH * FRAMES;             // scalar

  vq_prep<<<KCB, 256, 0, stream>>>(cb, cb16, csq, loss);
  vq_main<<<NVEC / 128, 256, 0, stream>>>(x, cb, cb16, csq, outq, codes, loss);
}
